// VectorQuantizer_47132971106721
// MI455X (gfx1250) — matrix operand order repaired and checked by reading, not compiled, not run on hardware
//
#include <hip/hip_runtime.h>
#include <hip/hip_bf16.h>

typedef __attribute__((ext_vector_type(16))) __bf16 v16bf;
typedef __attribute__((ext_vector_type(8)))  float  v8f;
typedef __attribute__((ext_vector_type(4)))  float  v4f;

#define D_DIM 256
#define K_DIM 1024
#define N_ROWS 65536   // 16*64*64

__device__ __forceinline__ __bf16 f2bf(float f) {
    // round-to-nearest-even f32 -> bf16 truncation
    unsigned u = __builtin_bit_cast(unsigned, f);
    u = (u + 0x7FFFu + ((u >> 16) & 1u)) >> 16;
    unsigned short s = (unsigned short)u;
    return __builtin_bit_cast(__bf16, s);
}

// ---------------------------------------------------------------------------
// Pack embeddings [D=256, K=1024] f32 into bf16 WMMA B-fragments.
// Fragment layout (dense 32x16 bf16 B, contiguous k order shared with A, not ISA table order): lane = {half(K16)}*16 + n,
// VGPR v holds K = 16*half + 2v, 2v+1.  packed[(c*8+s)*32 + lane] = v16bf.
// c = column tile (16 codes), s = dStep (32 of the 256 reduction dims).
// ---------------------------------------------------------------------------
__global__ void pack_b_kernel(const float* __restrict__ e, v16bf* __restrict__ packed) {
    int t    = blockIdx.x * blockDim.x + threadIdx.x;  // 0 .. 64*8*32-1
    int lane = t & 31;
    int s    = (t >> 5) & 7;
    int c    = t >> 8;
    int n    = lane & 15;
    int hf   = lane >> 4;
    int kbase = s * 32 + 16 * hf;
    int col   = c * 16 + n;
    v16bf frag;
#pragma unroll
    for (int v = 0; v < 8; ++v) {
#pragma unroll
        for (int p = 0; p < 2; ++p) {
            int k = kbase + 2 * v + p;            // reduction index (d)
            frag[2 * v + p] = f2bf(e[k * K_DIM + col]);
        }
    }
    packed[t] = frag;
}

// |e_k|^2 per code, coalesced over k.
__global__ void colnorm_kernel(const float* __restrict__ e, float* __restrict__ colNorm) {
    int k = blockIdx.x * blockDim.x + threadIdx.x;  // 0..1023
    float s = 0.f;
    for (int d = 0; d < D_DIM; ++d) { float v = e[d * K_DIM + k]; s += v * v; }
    colNorm[k] = s;
}

// eT[k*256 + d] = e[d*1024 + k]  (coalesced reads; 1 MB one-time)
__global__ void transpose_kernel(const float* __restrict__ e, float* __restrict__ eT) {
    int t = blockIdx.x * blockDim.x + threadIdx.x;  // 0..262143
    int d = t >> 10;
    int k = t & 1023;
    eT[k * D_DIM + d] = e[d * K_DIM + k];
}

// ---------------------------------------------------------------------------
// Main kernel: each wave32 owns 16 rows; sweeps 1024 codes with bf16 WMMA,
// tracks argmin of |x|^2 + |e|^2 - 2 x.e, gathers winning eT row to output.
// ---------------------------------------------------------------------------
__global__ void __launch_bounds__(256) vq_main_kernel(
    const float* __restrict__ x, const v16bf* __restrict__ packedB,
    const float* __restrict__ colNorm, const float* __restrict__ eT,
    float* __restrict__ out)
{
    int lane    = threadIdx.x & 31;
    int waveId  = blockIdx.x * 8 + (threadIdx.x >> 5);
    int rowTile = waveId * 16;

    int m  = lane & 15;
    int hf = lane >> 4;
    int row = rowTile + m;

    // Load A fragments (16x32 bf16, contiguous layout matching B: VGPR v -> K=16*hf+2v, 2v+1)
    // and accumulate this lane's share of |x_row|^2 in f32.
    v16bf a[8];
    float pnorm = 0.f;
#pragma unroll
    for (int s = 0; s < 8; ++s) {
        const float* xr = x + (long)row * D_DIM + s * 32;
#pragma unroll
        for (int v = 0; v < 8; ++v) {
            int k = 2 * v + 16 * hf;
            float f0 = xr[k];
            float f1 = xr[k + 1];
            pnorm += f0 * f0 + f1 * f1;
            a[s][2 * v]     = f2bf(f0);
            a[s][2 * v + 1] = f2bf(f1);
        }
    }
    // lanes l and l^16 jointly hold all 32 k's per step -> full row norm
    float rowNorm = pnorm + __shfl_xor(pnorm, 16, 32);
    // C/D layout: element v <-> row m = v + 8*hf; fetch those rows' norms
    float rn[8];
#pragma unroll
    for (int v = 0; v < 8; ++v)
        rn[v] = __shfl(rowNorm, v + 8 * hf, 32);

    float bestD[8];
    int   bestN[8];
#pragma unroll
    for (int v = 0; v < 8; ++v) { bestD[v] = 3.4e38f; bestN[v] = 0; }

    int nsub = lane & 15;
    for (int c = 0; c < 64; ++c) {
        v8f acc = {};
        const v16bf* bp = packedB + (c * 8) * 32 + lane;
#pragma unroll
        for (int s = 0; s < 8; ++s) {
            v16bf b = bp[s * 32];
            acc = __builtin_amdgcn_wmma_f32_16x16x32_bf16(
                false, a[s], false, b, (short)0, acc, false, false);
        }
        float cn = colNorm[c * 16 + nsub];
        int   n  = c * 16 + nsub;
#pragma unroll
        for (int v = 0; v < 8; ++v) {
            float dist = rn[v] + cn - 2.0f * acc[v];
            if (dist < bestD[v]) { bestD[v] = dist; bestN[v] = n; }  // first-min per lane
        }
    }

    // argmin across the 16 lanes of each half (xor<16 never crosses halves)
#pragma unroll
    for (int off = 1; off < 16; off <<= 1) {
#pragma unroll
        for (int v = 0; v < 8; ++v) {
            float od = __shfl_xor(bestD[v], off, 32);
            int   on = __shfl_xor(bestN[v], off, 32);
            if (od < bestD[v] || (od == bestD[v] && on < bestN[v])) {
                bestD[v] = od; bestN[v] = on;
            }
        }
    }

    // Decode: out[row] = eT[bestIdx]. Half 0 writes rows rowTile..+7,
    // half 1 rows rowTile+8..+15. 16 lanes x float4 contiguous per j.
    int q = lane & 15;
#pragma unroll
    for (int v = 0; v < 8; ++v) {
        int r   = rowTile + 8 * hf + v;
        int idx = bestN[v];
        const v4f* src = (const v4f*)(eT + (long)idx * D_DIM);
        v4f*       dst = (v4f*)(out + (long)r * D_DIM);
#pragma unroll
        for (int j = 0; j < 4; ++j)
            dst[j * 16 + q] = src[j * 16 + q];
    }
}

extern "C" void kernel_launch(void* const* d_in, const int* in_sizes, int n_in,
                              void* d_out, int out_size, void* d_ws, size_t ws_size,
                              hipStream_t stream) {
    const float* x = (const float*)d_in[0];        // [16,64,64,256] f32
    const float* e = (const float*)d_in[1];        // [256,1024] f32
    float* out     = (float*)d_out;                // [16,64,64,256] f32

    char* ws = (char*)d_ws;
    v16bf* packedB = (v16bf*)ws;                                   // 512 KB
    float* eT      = (float*)(ws + (512u << 10));                  // 1 MB
    float* colNorm = (float*)(ws + (512u << 10) + (1u << 20));     // 4 KB

    pack_b_kernel <<<  64, 256, 0, stream>>>(e, packedB);
    colnorm_kernel<<<   4, 256, 0, stream>>>(e, colNorm);
    transpose_kernel<<<1024, 256, 0, stream>>>(e, eT);
    vq_main_kernel<<< 512, 256, 0, stream>>>(x, packedB, colNorm, eT, out);
}
